// MSARowAttention_10849087390540
// MI455X (gfx1250) — compile-verified, matrix operand order checked by reading, not run on hardware
//
#include <hip/hip_runtime.h>
#include <math.h>

// ---- CDNA5 WMMA types ----
typedef __attribute__((ext_vector_type(16))) _Float16 v16h;
typedef __attribute__((ext_vector_type(8)))  _Float16 v8h;
typedef __attribute__((ext_vector_type(4)))  _Float16 v4h;
typedef __attribute__((ext_vector_type(8)))  float    v8f;

union V16H { v16h v; v8h h[2]; };

__device__ __forceinline__ v8f wmma_f16(const V16H& a, const V16H& b, v8f c) {
  // D = A(16x32 f16) * B(32x16 f16) + C(16x16 f32)
  return __builtin_amdgcn_wmma_f32_16x16x32_f16(false, a.v, false, b.v,
                                                (short)0, c, false, false);
}

// Problem constants: B=1, N=32, L=512, D=256, H=8, hd=32
#define SEQ_N   32
#define LEN_L   512
#define DIM_D   256
#define HEADS   8
#define HD      32
#define MROWS   (SEQ_N * LEN_L)   // 16384
#define QKV_N   (3 * DIM_D)       // 768

// ------- Kernel 0: weight convert + transpose: [k][n] f32 -> [n][k] f16 -----
__global__ __launch_bounds__(256) void cvt_tr_kernel(const float* __restrict__ src,
                                                     _Float16* __restrict__ dst,
                                                     int K, int N) {
  // out linear index over [n][k]; output writes coalesced in k
  int i = blockIdx.x * blockDim.x + threadIdx.x;
  if (i >= K * N) return;
  int k = i % K, n = i / K;
  dst[i] = (_Float16)src[k * N + n];
}

// ---------------- Kernel 0b: pair_bias transpose [q][k][h] -> [h][q][k] ----
__global__ __launch_bounds__(256) void bias_tr_kernel(const float* __restrict__ pb,
                                                      float* __restrict__ pbt) {
  int i = blockIdx.x * blockDim.x + threadIdx.x;   // 0 .. 8*512*512-1
  int k = i & (LEN_L - 1);
  int q = (i >> 9) & (LEN_L - 1);
  int h = i >> 18;
  pbt[i] = pb[(q * LEN_L + k) * HEADS + h];
}

// ---------------- Kernel 1: LayerNorm -> f16 ----------------
__global__ __launch_bounds__(256) void ln_kernel(const float* __restrict__ msa,
                                                 const float* __restrict__ lw,
                                                 const float* __restrict__ lb,
                                                 _Float16* __restrict__ mh) {
  __shared__ float red[256];
  const int row = blockIdx.x;
  const int t = threadIdx.x;
  float x = msa[row * DIM_D + t];
  red[t] = x; __syncthreads();
  for (int s = 128; s > 0; s >>= 1) { if (t < s) red[t] += red[t + s]; __syncthreads(); }
  float mu = red[0] * (1.0f / DIM_D);
  __syncthreads();
  float dx = x - mu;
  red[t] = dx * dx; __syncthreads();
  for (int s = 128; s > 0; s >>= 1) { if (t < s) red[t] += red[t + s]; __syncthreads(); }
  float var = red[0] * (1.0f / DIM_D);
  float y = dx * rsqrtf(var + 1e-5f) * lw[t] + lb[t];
  mh[row * DIM_D + t] = (_Float16)y;
}

// ============ shared GEMM body (double-buffered LDS ping-pong) ============
// Block tile 128x64, 8 waves, wave = 16 rows x 64 cols (4 C frags), BK=32.
// A: [M][256] f16 row-major. BhT: [N][256] f16 (pre-transposed weights).
__device__ __forceinline__ void gemm_mainloop(const _Float16* __restrict__ Ah,
                                              const _Float16* __restrict__ BhT,
                                              int row0, int ntile,
                                              _Float16 (*As)[128][32],
                                              _Float16 (*BsT)[64][32],
                                              v8f acc[4]) {
  const int t = threadIdx.x, wid = t >> 5, lane = t & 31;
  const int lrow = lane & 15, lhi = lane >> 4;

  const int ar  = t >> 1, aseg = (t & 1) * 16;   // A staging: 16 halves/thread
  const int bn  = t & 63, bks  = (t >> 6) * 8;   // B staging: 8 halves/thread

  v8h ra0, ra1, rb;
  auto gload = [&](int kt) {
    const _Float16* srcA = Ah + (row0 + ar) * DIM_D + kt + aseg;
    ra0 = *(const v8h*)srcA;
    ra1 = *(const v8h*)(srcA + 8);
    rb  = *(const v8h*)(BhT + (ntile + bn) * DIM_D + kt + bks);  // contiguous
  };
  auto sstore = [&](int buf) {
    *(v8h*)&As[buf][ar][aseg]     = ra0;
    *(v8h*)&As[buf][ar][aseg + 8] = ra1;
    *(v8h*)&BsT[buf][bn][bks]     = rb;
  };

  gload(0);
  sstore(0);
  __syncthreads();

  for (int step = 0; step < DIM_D / 32; ++step) {
    const int buf = step & 1;
    const bool more = (step + 1) < DIM_D / 32;
    if (more) gload((step + 1) * 32);            // overlap VMEM with WMMA

    V16H a;
    a.h[0] = *(const v8h*)&As[buf][wid * 16 + lrow][lhi * 8];
    a.h[1] = *(const v8h*)&As[buf][wid * 16 + lrow][lhi * 8 + 16];
    V16H b[4];
    #pragma unroll
    for (int nf = 0; nf < 4; ++nf) {             // batch all ds loads first
      b[nf].h[0] = *(const v8h*)&BsT[buf][nf * 16 + lrow][lhi * 8];
      b[nf].h[1] = *(const v8h*)&BsT[buf][nf * 16 + lrow][lhi * 8 + 16];
    }
    #pragma unroll
    for (int nf = 0; nf < 4; ++nf)               // 4 back-to-back WMMAs
      acc[nf] = wmma_f16(a, b[nf], acc[nf]);

    if (more) sstore(buf ^ 1);
    __syncthreads();
  }
}

// ---------------- Kernel 2: QKV GEMM (16384x256 x 256x768) ----------------
__global__ __launch_bounds__(256) void qkv_gemm_kernel(const _Float16* __restrict__ Ah,
                                                       const _Float16* __restrict__ BhT,
                                                       const float* __restrict__ bq,
                                                       _Float16* __restrict__ Qh,
                                                       _Float16* __restrict__ Kh,
                                                       _Float16* __restrict__ Vh) {
  __shared__ __align__(16) _Float16 As[2][128][32];   // 16 KB
  __shared__ __align__(16) _Float16 BsT[2][64][32];   //  8 KB
  const int row0  = blockIdx.x * 128;
  const int ntile = blockIdx.y * 64;
  const int t = threadIdx.x, wid = t >> 5, lane = t & 31;
  const int lrow = lane & 15, lhi = lane >> 4;

  v8f acc[4] = {};
  gemm_mainloop(Ah, BhT, row0, ntile, As, BsT, acc);

  // Epilogue: + bias, scatter into Q/K/V [n][h][l][hd] f16
  #pragma unroll
  for (int nf = 0; nf < 4; ++nf) {
    #pragma unroll
    for (int r = 0; r < 8; ++r) {
      int col = ntile + nf * 16 + lrow;
      int row = row0 + wid * 16 + r + 8 * lhi;
      float v = acc[nf][r] + bq[col];
      int which = col >> 8, d = col & 255, hh = d >> 5, hdi = d & 31;
      _Float16* dst = (which == 0) ? Qh : ((which == 1) ? Kh : Vh);
      int n = row >> 9, l = row & 511;
      dst[((n * HEADS + hh) * LEN_L + l) * HD + hdi] = (_Float16)v;
    }
  }
}

// ---------------- Kernel 4: proj GEMM + bias + residual ----------------
__global__ __launch_bounds__(256) void proj_gemm_kernel(const _Float16* __restrict__ Ah,
                                                        const _Float16* __restrict__ BhT,
                                                        const float* __restrict__ bp,
                                                        const float* __restrict__ resid,
                                                        float* __restrict__ out) {
  __shared__ __align__(16) _Float16 As[2][128][32];
  __shared__ __align__(16) _Float16 BsT[2][64][32];
  const int row0  = blockIdx.x * 128;
  const int ntile = blockIdx.y * 64;
  const int t = threadIdx.x, wid = t >> 5, lane = t & 31;
  const int lrow = lane & 15, lhi = lane >> 4;

  v8f acc[4] = {};
  gemm_mainloop(Ah, BhT, row0, ntile, As, BsT, acc);

  #pragma unroll
  for (int nf = 0; nf < 4; ++nf) {
    #pragma unroll
    for (int r = 0; r < 8; ++r) {
      int col = ntile + nf * 16 + lrow;
      int row = row0 + wid * 16 + r + 8 * lhi;
      out[row * DIM_D + col] = acc[nf][r] + bp[col] + resid[row * DIM_D + col];
    }
  }
}

// ---------------- Kernel 3: flash attention per (n,h) ----------------
// Block: 128 queries (8 waves x 16), loops key chunks of 32 (double-buffered).
__global__ __launch_bounds__(256) void attn_kernel(const _Float16* __restrict__ Qh,
                                                   const _Float16* __restrict__ Kh,
                                                   const _Float16* __restrict__ Vh,
                                                   const float* __restrict__ pbt, // [h][q][k]
                                                   _Float16* __restrict__ AO) {
  __shared__ __align__(16) _Float16 Ks[2][32][32];     // [key][d]
  __shared__ __align__(16) _Float16 Vt[2][32][32];     // [d][key]
  __shared__ __align__(16) _Float16 Ps[8][16][32];     // per-wave P [q][key]
  const int nh = blockIdx.y;                 // n*8 + h
  const int n = nh >> 3, h = nh & 7;
  const int q0 = blockIdx.x * 128;
  const int t = threadIdx.x, wid = t >> 5, lane = t & 31;
  const int lrow = lane & 15, lhi = lane >> 4;

  const _Float16* Qb = Qh + (size_t)nh * LEN_L * HD;
  const _Float16* Kb = Kh + (size_t)nh * LEN_L * HD;
  const _Float16* Vb = Vh + (size_t)nh * LEN_L * HD;
  const float* biasT = pbt + (size_t)h * LEN_L * LEN_L;   // [q][k] contiguous

  V16H aq;
  {
    int qrow = q0 + wid * 16 + lrow;
    aq.h[0] = *(const v8h*)(Qb + qrow * HD + lhi * 8);
    aq.h[1] = *(const v8h*)(Qb + qrow * HD + lhi * 8 + 16);
  }
  float rmax[8], rsum[8];
  v8f o0 = {}, o1 = {};
  #pragma unroll
  for (int r = 0; r < 8; ++r) { rmax[r] = -1e30f; rsum[r] = 0.0f; }
  const float scale = 0.17677669529663687f;  // 1/sqrt(32)

  const int skey = t >> 3, sds = (t & 7) * 4;  // staging: 4 halves/thread each
  v4h kr, vr;
  auto gload = [&](int kc) {
    kr = *(const v4h*)(Kb + (kc + skey) * HD + sds);
    vr = *(const v4h*)(Vb + (kc + skey) * HD + sds);
  };
  auto sstore = [&](int buf) {
    *(v4h*)&Ks[buf][skey][sds] = kr;
    Vt[buf][sds + 0][skey] = vr[0]; Vt[buf][sds + 1][skey] = vr[1];
    Vt[buf][sds + 2][skey] = vr[2]; Vt[buf][sds + 3][skey] = vr[3];
  };

  gload(0);
  sstore(0);
  __syncthreads();

  for (int kc = 0; kc < LEN_L; kc += 32) {
    const int buf = (kc >> 5) & 1;
    const bool more = (kc + 32) < LEN_L;
    if (more) {
      gload(kc + 32);                                      // overlap with compute
      __builtin_prefetch(&biasT[(q0 + wid * 16 + lhi * 8) * LEN_L + kc + 32 + lrow], 0, 0);
    }

    // scores: S(16x32) = Q(16x32) * K^T  (B frag [d][key] == Ks row-major)
    V16H bk0, bk1;
    bk0.h[0] = *(const v8h*)&Ks[buf][lrow][lhi * 8];
    bk0.h[1] = *(const v8h*)&Ks[buf][lrow][lhi * 8 + 16];
    bk1.h[0] = *(const v8h*)&Ks[buf][16 + lrow][lhi * 8];
    bk1.h[1] = *(const v8h*)&Ks[buf][16 + lrow][lhi * 8 + 16];
    v8f s0 = {}, s1 = {};
    s0 = wmma_f16(aq, bk0, s0);
    s1 = wmma_f16(aq, bk1, s1);

    // online softmax (fp32); C layout: col=lane&15, row=r+8*(lane>>4)
    #pragma unroll
    for (int r = 0; r < 8; ++r) {
      int q = q0 + wid * 16 + r + 8 * lhi;
      float b0 = biasT[q * LEN_L + kc + lrow];        // coalesced across lanes
      float b1 = biasT[q * LEN_L + kc + 16 + lrow];
      float sv0 = s0[r] * scale + b0;
      float sv1 = s1[r] * scale + b1;
      float mc = fmaxf(sv0, sv1);
      #pragma unroll
      for (int off = 1; off < 16; off <<= 1) mc = fmaxf(mc, __shfl_xor(mc, off, 16));
      float mnew = fmaxf(rmax[r], mc);
      float alpha = __expf(rmax[r] - mnew);
      rmax[r] = mnew;
      float p0 = __expf(sv0 - mnew);
      float p1 = __expf(sv1 - mnew);
      float ps = p0 + p1;
      #pragma unroll
      for (int off = 1; off < 16; off <<= 1) ps += __shfl_xor(ps, off, 16);
      rsum[r] = rsum[r] * alpha + ps;
      o0[r] *= alpha; o1[r] *= alpha;
      // C-layout -> A-layout transpose via per-wave LDS
      Ps[wid][r + 8 * lhi][lrow]      = (_Float16)p0;
      Ps[wid][r + 8 * lhi][16 + lrow] = (_Float16)p1;
    }

    // O(16x32) += P(16x32) * V(32x32); B frag [key][hd] -> read Vt[hd][key]
    V16H ap, bv0, bv1;
    ap.h[0] = *(const v8h*)&Ps[wid][lrow][lhi * 8];
    ap.h[1] = *(const v8h*)&Ps[wid][lrow][lhi * 8 + 16];
    bv0.h[0] = *(const v8h*)&Vt[buf][lrow][lhi * 8];
    bv0.h[1] = *(const v8h*)&Vt[buf][lrow][lhi * 8 + 16];
    bv1.h[0] = *(const v8h*)&Vt[buf][16 + lrow][lhi * 8];
    bv1.h[1] = *(const v8h*)&Vt[buf][16 + lrow][lhi * 8 + 16];
    o0 = wmma_f16(ap, bv0, o0);
    o1 = wmma_f16(ap, bv1, o1);

    if (more) sstore(buf ^ 1);
    __syncthreads();
  }

  // normalize + store to attn_out [n][l][h*32+hd] f16
  #pragma unroll
  for (int r = 0; r < 8; ++r) {
    int q = q0 + wid * 16 + r + 8 * lhi;
    float inv = 1.0f / rsum[r];
    size_t base = ((size_t)n * LEN_L + q) * DIM_D + h * HD;
    AO[base + lrow]      = (_Float16)(o0[r] * inv);
    AO[base + 16 + lrow] = (_Float16)(o1[r] * inv);
  }
}

// ---------------- launch ----------------
extern "C" void kernel_launch(void* const* d_in, const int* in_sizes, int n_in,
                              void* d_out, int out_size, void* d_ws, size_t ws_size,
                              hipStream_t stream) {
  const float* msa   = (const float*)d_in[0];
  const float* pbias = (const float*)d_in[1];
  const float* lnw   = (const float*)d_in[2];
  const float* lnb   = (const float*)d_in[3];
  const float* wqkv  = (const float*)d_in[4];
  const float* bqkv  = (const float*)d_in[5];
  const float* wproj = (const float*)d_in[6];
  const float* bproj = (const float*)d_in[7];
  float* out = (float*)d_out;

  char* ws = (char*)d_ws;
  // workspace layout (bytes), all offsets 16B-aligned
  _Float16* WqkvT  = (_Float16*)(ws + 0);          // [768][256] f16 = 393216
  _Float16* WprojT = (_Float16*)(ws + 393216);     // [256][256] f16 = 131072
  _Float16* Mh     = (_Float16*)(ws + 524288);     // 16384*256*2 = 8388608
  _Float16* Qh     = (_Float16*)(ws + 8912896);    // 8388608
  _Float16* Kh     = (_Float16*)(ws + 17301504);   // 8388608
  _Float16* Vh     = (_Float16*)(ws + 25690112);   // 8388608
  _Float16* AOh    = (_Float16*)(ws + 34078720);   // 8388608
  float*    PbT    = (float*)   (ws + 42467328);   // 8*512*512*4 = 8388608 (total ~48.5 MB)

  cvt_tr_kernel<<<(DIM_D * QKV_N + 255) / 256, 256, 0, stream>>>(wqkv, WqkvT, DIM_D, QKV_N);
  cvt_tr_kernel<<<(DIM_D * DIM_D + 255) / 256, 256, 0, stream>>>(wproj, WprojT, DIM_D, DIM_D);
  bias_tr_kernel<<<(HEADS * LEN_L * LEN_L) / 256, 256, 0, stream>>>(pbias, PbT);
  ln_kernel<<<MROWS, 256, 0, stream>>>(msa, lnw, lnb, Mh);
  qkv_gemm_kernel<<<dim3(MROWS / 128, QKV_N / 64), 256, 0, stream>>>(Mh, WqkvT, bqkv, Qh, Kh, Vh);
  attn_kernel<<<dim3(LEN_L / 128, SEQ_N * HEADS), 256, 0, stream>>>(Qh, Kh, Vh, PbT, AOh);
  proj_gemm_kernel<<<dim3(MROWS / 128, DIM_D / 64), 256, 0, stream>>>(AOh, WprojT, bproj, msa, out);
}
